// Bi_Dir_FeatureExtractor_35854386987567
// MI455X (gfx1250) — compile-verified, matrix operand order checked by reading, not run on hardware
//
#include <hip/hip_runtime.h>
#include <hip/hip_bf16.h>
#include <math.h>

// ---------------------------------------------------------------------------
// CDNA5 (gfx1250) implementation of the bi-directional feature extractor.
// All 3x3 convs (pre towers, ext convs, zero convs) run as implicit-GEMM
// through v_wmma_f32_16x16x32_bf16 (wave32). Each wave computes a
// 16(pixel) x NT*16(channel) tile: one im2col A fragment is reused across
// NT B fragments (staged in LDS), amortizing the gather cost and issuing
// NT independent WMMAs per K-step. Splatting uses global_atomic_add_f32.
// ---------------------------------------------------------------------------

typedef __attribute__((ext_vector_type(16))) __bf16 v16bf;
typedef __attribute__((ext_vector_type(8)))  float  v8f;

#define DEV static __device__ __forceinline__
#define CDIV(a, b) (((a) + (b) - 1) / (b))

DEV unsigned short f32_to_bf16u(float f) {
  union { float f; unsigned u; } v; v.f = f;
  unsigned r = 0x7FFFu + ((v.u >> 16) & 1u);      // round-to-nearest-even
  return (unsigned short)((v.u + r) >> 16);
}

DEV int clampi(int v, int lo, int hi) { return v < lo ? lo : (v > hi ? hi : v); }

// ---------------------------------------------------------------------------
// Weight packing: OIHW fp32 -> [O][Kpad] bf16 im2col layout (k = ci*9+ky*3+kx)
// ---------------------------------------------------------------------------
__global__ void pack_w_kernel(const float* __restrict__ w, unsigned short* __restrict__ out,
                              int O, int I, int ksz, int Kpad) {
  int idx = blockIdx.x * blockDim.x + threadIdx.x;
  int total = O * Kpad;
  if (idx >= total) return;
  int o = idx / Kpad, k = idx - o * Kpad;
  int Kt = I * ksz * ksz;
  float v = (k < Kt) ? w[(size_t)o * Kt + k] : 0.f;
  out[idx] = f32_to_bf16u(v);
}

// ---------------------------------------------------------------------------
// Extract 3 channels from local_conditions [8][6][512][512] -> bf16 [8][3][512][512]
// ---------------------------------------------------------------------------
__global__ void slice_bf16_kernel(const float* __restrict__ lc, unsigned short* __restrict__ dst,
                                  int cbase, int n) {
  int i = blockIdx.x * blockDim.x + threadIdx.x;
  if (i >= n) return;
  const int HW = 512 * 512;
  int b = i / (3 * HW);
  int rem = i - b * 3 * HW;
  int c = rem / HW;
  int p = rem - c * HW;
  dst[i] = f32_to_bf16u(lc[((size_t)(b * 6 + cbase + c)) * HW + p]);
}

// ---------------------------------------------------------------------------
// Implicit-GEMM conv with bf16 WMMA, fused bias+SiLU.
// One wave = 16 output pixels x NT*16 output channels. K = Ci*ksz^2 pad 32.
//   A layout (16-bit A 16x32):   M = lane%16,  K half selected by lane/16
//   B layout (16-bit B 32x16):   N = lane%16,  K half selected by lane/16
//   D layout (f32 16x16):        M = r + 8*(lane/16), N = lane%16
// ---------------------------------------------------------------------------
template <int NT>
__global__ void __launch_bounds__(256)
conv_wmma_kernel(const unsigned short* __restrict__ xin,   // bf16 [B][Ci][Hi][Wi]
                 const unsigned short* __restrict__ wpk,   // bf16 [O][Kpad]
                 const float* __restrict__ bias,           // [O]
                 float* __restrict__ yf32,                 // nullable [B][O][Ho][Wo]
                 unsigned short* __restrict__ ybf16,       // nullable [B][O][Ho][Wo]
                 int Ci, int Hi, int Wi, int O, int Ho, int Wo,
                 int stride, int pad, int ksz, int Ktrue, int Kpad, int silu) {
  __shared__ __align__(16) unsigned short wlds[NT * 16 * 32];   // [n][k] tile
  const int lane = threadIdx.x & 31;
  const int wave = threadIdx.x >> 5;
  const int hi   = lane >> 4;
  const int lm   = lane & 15;
  const int npix = Ho * Wo;
  const int pix0 = (blockIdx.x * 8 + wave) * 16;
  const int n0   = blockIdx.y * (16 * NT);
  const int b    = blockIdx.z;

  const int  m   = pix0 + lm;
  const bool mv  = (m < npix);
  const int  mm  = mv ? m : 0;
  const int  oy  = mm / Wo;
  const int  ox  = mm - oy * Wo;
  const int  khw = ksz * ksz;

  v8f acc[NT];
#pragma unroll
  for (int t = 0; t < NT; ++t) acc[t] = (v8f){};

  for (int k0 = 0; k0 < Kpad; k0 += 32) {
    // Stage NT*16(N) x 32(K) bf16 weight tile into LDS (shared by all 8 waves).
    for (int t = threadIdx.x; t < NT * 512; t += 256) {
      int nn = t >> 5, kk = t & 31;
      wlds[t] = wpk[(size_t)(n0 + nn) * Kpad + (k0 + kk)];
    }
    __syncthreads();
    if (k0 + 32 < Kpad)  // pull next K-step weight tile toward L2/L0
      __builtin_prefetch(&wpk[(size_t)n0 * Kpad + k0 + 32], 0, 0);

    // A fragment: per-lane im2col gather (bf16), reused across all NT tiles.
    union { unsigned short u16[16]; v16bf v; } afr;
#pragma unroll
    for (int v = 0; v < 8; ++v) {
      int kb = ((v < 4) ? 0 : 16) + hi * 8 + (v & 3) * 2;
#pragma unroll
      for (int e = 0; e < 2; ++e) {
        int kk = k0 + kb + e;
        unsigned short val = 0;
        if (mv && kk < Ktrue) {
          int ci = kk / khw;
          int r  = kk - ci * khw;
          int ky = r / ksz;
          int kx = r - ky * ksz;
          int iy = oy * stride - pad + ky;
          int ix = ox * stride - pad + kx;
          if ((unsigned)iy < (unsigned)Hi && (unsigned)ix < (unsigned)Wi)
            val = xin[((size_t)(b * Ci + ci) * Hi + iy) * Wi + ix];
        }
        afr.u16[v * 2 + e] = val;
      }
    }

    // NT independent B fragments / WMMAs (no D->A/B hazard between them).
    const unsigned int* l32 = reinterpret_cast<const unsigned int*>(wlds);
#pragma unroll
    for (int t = 0; t < NT; ++t) {
      union { unsigned int u32[8]; v16bf v; } bfr;
#pragma unroll
      for (int v = 0; v < 8; ++v) {
        int kb = ((v < 4) ? 0 : 16) + hi * 8 + (v & 3) * 2;
        bfr.u32[v] = l32[((t * 16 + lm) * 32 + kb) >> 1];
      }
      acc[t] = __builtin_amdgcn_wmma_f32_16x16x32_bf16(
          false, afr.v, false, bfr.v, (short)0, acc[t], false, false);
    }
    __syncthreads();
  }

  // Epilogue: bias + optional SiLU, write fp32 and/or bf16.
#pragma unroll
  for (int t = 0; t < NT; ++t) {
    int ch = n0 + t * 16 + lm;
    float bval = bias[ch];
#pragma unroll
    for (int r = 0; r < 8; ++r) {
      int pixel = pix0 + hi * 8 + r;
      if (pixel < npix) {
        float v = acc[t][r] + bval;
        if (silu) v = v / (1.f + __expf(-v));
        size_t oi = ((size_t)(b * O + ch)) * npix + pixel;
        if (yf32)  yf32[oi]  = v;
        if (ybf16) ybf16[oi] = f32_to_bf16u(v);
      }
    }
  }
}

// ---------------------------------------------------------------------------
// Bilinear resize 512 -> res with half-pixel centers, then * (res/512).
// cbase selects flow channels {0,1} (fwd) or {2,3} (bwd). out: [8][2][res][res]
// ---------------------------------------------------------------------------
__global__ void flow_resize_kernel(const float* __restrict__ flow, float* __restrict__ out,
                                   int cbase, int res) {
  int i = blockIdx.x * blockDim.x + threadIdx.x;
  int total = 8 * 2 * res * res;
  if (i >= total) return;
  int hw = res * res;
  int b = i / (2 * hw);
  int rem = i - b * 2 * hw;
  int c = rem / hw;
  int p = rem - c * hw;
  int y = p / res, x = p - y * res;
  const int S = 512;
  float scale = (float)S / (float)res;
  float sx = ((float)x + 0.5f) * scale - 0.5f;
  float sy = ((float)y + 0.5f) * scale - 0.5f;
  int x0 = (int)floorf(sx), y0 = (int)floorf(sy);
  float fx = sx - (float)x0, fy = sy - (float)y0;
  int x0c = clampi(x0, 0, S - 1), x1c = clampi(x0 + 1, 0, S - 1);
  int y0c = clampi(y0, 0, S - 1), y1c = clampi(y0 + 1, 0, S - 1);
  const float* src = flow + ((size_t)(b * 4 + cbase + c)) * S * S;
  float v = (1 - fx) * (1 - fy) * src[y0c * S + x0c] + fx * (1 - fy) * src[y0c * S + x1c] +
            (1 - fx) * fy       * src[y1c * S + x0c] + fx * fy       * src[y1c * S + x1c];
  out[i] = v * ((float)res / (float)S);
}

// ---------------------------------------------------------------------------
// Forward-backward consistency mask. fa/fb: [8][2][res][res]; mask: [8][res][res]
// ---------------------------------------------------------------------------
__global__ void occ_mask_kernel(const float* __restrict__ fa, const float* __restrict__ fb,
                                float* __restrict__ mask, int res) {
  int i = blockIdx.x * blockDim.x + threadIdx.x;
  int total = 8 * res * res;
  if (i >= total) return;
  int hw = res * res;
  int b = i / hw;
  int p = i - b * hw;
  int y = p / res, x = p - y * res;
  const float* A  = fa + (size_t)b * 2 * hw;
  const float* B0 = fb + (size_t)b * 2 * hw;
  const float* B1 = B0 + hw;
  float a0 = A[p], a1 = A[hw + p];
  float sx = (float)x + a0, sy = (float)y + a1;
  float x0 = floorf(sx), y0 = floorf(sy);
  float fx = sx - x0, fy = sy - y0;
  int xi0 = clampi((int)x0, 0, res - 1), xi1 = clampi((int)x0 + 1, 0, res - 1);
  int yi0 = clampi((int)y0, 0, res - 1), yi1 = clampi((int)y0 + 1, 0, res - 1);
  float w00 = (1 - fx) * (1 - fy), w01 = fx * (1 - fy), w10 = (1 - fx) * fy, w11 = fx * fy;
  float wb0 = w00 * B0[yi0 * res + xi0] + w01 * B0[yi0 * res + xi1] +
              w10 * B0[yi1 * res + xi0] + w11 * B0[yi1 * res + xi1];
  float wb1 = w00 * B1[yi0 * res + xi0] + w01 * B1[yi0 * res + xi1] +
              w10 * B1[yi1 * res + xi0] + w11 * B1[yi1 * res + xi1];
  float d   = (a0 + wb0) * (a0 + wb0) + (a1 + wb1) * (a1 + wb1);
  float thr = 0.01f * ((a0 * a0 + a1 * a1) + (wb0 * wb0 + wb1 * wb1)) + 0.5f;
  mask[i] = (d > thr) ? 1.f : 0.f;
}

// ---------------------------------------------------------------------------
// metric (1x1 conv) -> clip -> exp -> * (1 - mask).  Z: [8][res][res]
// ---------------------------------------------------------------------------
__global__ void metric_z_kernel(const float* __restrict__ feat, const float* __restrict__ mw,
                                const float* __restrict__ mb, const float* __restrict__ mask,
                                float* __restrict__ Z, int C, int res) {
  int i = blockIdx.x * blockDim.x + threadIdx.x;
  int total = 8 * res * res;
  if (i >= total) return;
  int hw = res * res;
  int b = i / hw;
  int p = i - b * hw;
  const float* f = feat + ((size_t)b * C) * hw + p;
  float acc = mb[0];
  for (int c = 0; c < C; ++c) acc += mw[c] * f[(size_t)c * hw];
  acc = fminf(fmaxf(acc, -20.f), 20.f);
  Z[i] = __expf(acc) * (1.f - mask[i]);
}

__global__ void zero_f32_kernel(float* __restrict__ p, int n) {
  int i = blockIdx.x * blockDim.x + threadIdx.x;
  if (i < n) p[i] = 0.f;
}

// ---------------------------------------------------------------------------
// Softsplat forward scatter: acc[b][c][y][x] += w * (c<C ? Z*feat : Z)
// acc: [8][C+1][res][res] -- uses global_atomic_add_f32
// ---------------------------------------------------------------------------
__global__ void splat_kernel(const float* __restrict__ feat, const float* __restrict__ Z,
                             const float* __restrict__ flow, float* __restrict__ acc,
                             int C, int res) {
  int i = blockIdx.x * blockDim.x + threadIdx.x;
  int hw = res * res;
  int total = 8 * (C + 1) * hw;
  if (i >= total) return;
  int b = i / ((C + 1) * hw);
  int rem = i - b * (C + 1) * hw;
  int c = rem / hw;
  int p = rem - c * hw;
  int y = p / res, x = p - y * res;
  const float* fl = flow + (size_t)b * 2 * hw;
  float tx = (float)x + fl[p];
  float ty = (float)y + fl[hw + p];
  int x0 = (int)floorf(tx), y0 = (int)floorf(ty);
  float fx = tx - (float)x0, fy = ty - (float)y0;
  float z = Z[b * hw + p];
  float val = (c < C) ? z * feat[((size_t)(b * C + c)) * hw + p] : z;
  float w00 = (1 - fx) * (1 - fy), w01 = fx * (1 - fy), w10 = (1 - fx) * fy, w11 = fx * fy;
  float* out = acc + ((size_t)(b * (C + 1) + c)) * hw;
  if ((unsigned)x0 < (unsigned)res && (unsigned)y0 < (unsigned)res)
    atomicAdd(&out[y0 * res + x0], val * w00);
  if ((unsigned)(x0 + 1) < (unsigned)res && (unsigned)y0 < (unsigned)res)
    atomicAdd(&out[y0 * res + x0 + 1], val * w01);
  if ((unsigned)x0 < (unsigned)res && (unsigned)(y0 + 1) < (unsigned)res)
    atomicAdd(&out[(y0 + 1) * res + x0], val * w10);
  if ((unsigned)(x0 + 1) < (unsigned)res && (unsigned)(y0 + 1) < (unsigned)res)
    atomicAdd(&out[(y0 + 1) * res + x0 + 1], val * w11);
}

// ---------------------------------------------------------------------------
// Confidence fusion + hole fill; writes bf16 fused features for the zero conv.
// ---------------------------------------------------------------------------
__global__ void fuse_kernel(const float* __restrict__ accF, const float* __restrict__ accB,
                            const float* __restrict__ occF, const float* __restrict__ occB,
                            unsigned short* __restrict__ fused, int C, int res) {
  int i = blockIdx.x * blockDim.x + threadIdx.x;
  int hw = res * res;
  int total = 8 * C * hw;
  if (i >= total) return;
  int b = i / (C * hw);
  int rem = i - b * C * hw;
  int c = rem / hw;
  int p = rem - c * hw;
  const float* aF = accF + ((size_t)b * (C + 1)) * hw;
  const float* aB = accB + ((size_t)b * (C + 1)) * hw;
  float denF = aF[(size_t)C * hw + p];
  float denB = aB[(size_t)C * hw + p];
  float wf = aF[(size_t)c * hw + p] / (denF + 1e-7f);
  float wl = aB[(size_t)c * hw + p] / (denB + 1e-7f);
  float cf = fmaxf(denF, 0.f), cb = fmaxf(denB, 0.f);
  float s = cf + cb + 1e-6f;
  float v = (cf / s) * wf + (cb / s) * wl;
  if (occF[b * hw + p] + occB[b * hw + p] > 1.5f) v = 0.5f * (wf + wl);
  fused[i] = f32_to_bf16u(v);
}

// ---------------------------------------------------------------------------
// Host orchestration
// ---------------------------------------------------------------------------
struct ConvDims { int Ho, Wo, Ktrue, Kpad; };
static ConvDims conv_dims(int Hi, int Wi, int Ci, int ksz, int stride, int pad) {
  ConvDims d;
  d.Ho = (Hi + 2 * pad - ksz) / stride + 1;
  d.Wo = (Wi + 2 * pad - ksz) / stride + 1;
  d.Ktrue = Ci * ksz * ksz;
  d.Kpad = CDIV(d.Ktrue, 32) * 32;
  return d;
}

static void run_conv(hipStream_t s, const unsigned short* xin, const float* w, const float* bias,
                     unsigned short* wpk, float* yf, unsigned short* yb,
                     int Ci, int Hi, int Wi, int O, int ksz, int stride, int pad, int silu) {
  ConvDims d = conv_dims(Hi, Wi, Ci, ksz, stride, pad);
  int nw = O * d.Kpad;
  pack_w_kernel<<<CDIV(nw, 256), 256, 0, s>>>(w, wpk, O, Ci, ksz, d.Kpad);
  int nt = (O % 64 == 0) ? 4 : (O % 32 == 0) ? 2 : 1;
  dim3 grid(CDIV(d.Ho * d.Wo, 128), O / (16 * nt), 8);
  if (nt == 4)
    conv_wmma_kernel<4><<<grid, 256, 0, s>>>(xin, wpk, bias, yf, yb, Ci, Hi, Wi, O,
                                             d.Ho, d.Wo, stride, pad, ksz, d.Ktrue, d.Kpad, silu);
  else if (nt == 2)
    conv_wmma_kernel<2><<<grid, 256, 0, s>>>(xin, wpk, bias, yf, yb, Ci, Hi, Wi, O,
                                             d.Ho, d.Wo, stride, pad, ksz, d.Ktrue, d.Kpad, silu);
  else
    conv_wmma_kernel<1><<<grid, 256, 0, s>>>(xin, wpk, bias, yf, yb, Ci, Hi, Wi, O,
                                             d.Ho, d.Wo, stride, pad, ksz, d.Ktrue, d.Kpad, silu);
}

extern "C" void kernel_launch(void* const* d_in, const int* in_sizes, int n_in,
                              void* d_out, int out_size, void* d_ws, size_t ws_size,
                              hipStream_t stream) {
  (void)in_sizes; (void)n_in; (void)out_size; (void)ws_size;
  const float* lc   = (const float*)d_in[0];
  const float* flow = (const float*)d_in[1];

  // params flattened by sorted dict keys: ext_first, ext_last, metric,
  // pre_first, pre_last, zero; each leaf dict sorted: b, w.
  int pi = 2;
  const float *eFb[4], *eFw[4], *eLb[4], *eLw[4], *mb[4], *mw[4];
  const float *pFb[5], *pFw[5], *pLb[5], *pLw[5], *zb[4], *zw[4];
  for (int j = 0; j < 4; ++j) { eFb[j] = (const float*)d_in[pi++]; eFw[j] = (const float*)d_in[pi++]; }
  for (int j = 0; j < 4; ++j) { eLb[j] = (const float*)d_in[pi++]; eLw[j] = (const float*)d_in[pi++]; }
  for (int j = 0; j < 4; ++j) { mb[j]  = (const float*)d_in[pi++]; mw[j]  = (const float*)d_in[pi++]; }
  for (int j = 0; j < 5; ++j) { pFb[j] = (const float*)d_in[pi++]; pFw[j] = (const float*)d_in[pi++]; }
  for (int j = 0; j < 5; ++j) { pLb[j] = (const float*)d_in[pi++]; pLw[j] = (const float*)d_in[pi++]; }
  for (int j = 0; j < 4; ++j) { zb[j]  = (const float*)d_in[pi++]; zw[j]  = (const float*)d_in[pi++]; }

  char* ws = (char*)d_ws;
  size_t off = 0;
  auto alloc = [&](size_t bytes) -> void* {
    off = (off + 255) & ~(size_t)255;
    void* p = ws + off;
    off += bytes;
    return p;
  };

  const int Bn = 8;
  // --- input slices -> bf16 ---
  unsigned short* firstbf = (unsigned short*)alloc((size_t)Bn * 3 * 512 * 512 * 2);
  unsigned short* lastbf  = (unsigned short*)alloc((size_t)Bn * 3 * 512 * 512 * 2);
  {
    int n = Bn * 3 * 512 * 512;
    slice_bf16_kernel<<<CDIV(n, 256), 256, 0, stream>>>(lc, firstbf, 3, n);
    slice_bf16_kernel<<<CDIV(n, 256), 256, 0, stream>>>(lc, lastbf, 0, n);
  }

  // --- pre towers (bf16 intermediates only) ---
  const int preO[5] = {16, 32, 32, 64, 64};
  const int preI[5] = {3, 16, 32, 32, 64};
  const int preS[5] = {1, 2, 1, 2, 1};
  unsigned short* curF = firstbf;
  unsigned short* curL = lastbf;
  int H = 512, W = 512;
  for (int l = 0; l < 5; ++l) {
    ConvDims d = conv_dims(H, W, preI[l], 3, preS[l], 1);
    unsigned short* wpF  = (unsigned short*)alloc((size_t)preO[l] * d.Kpad * 2);
    unsigned short* wpL  = (unsigned short*)alloc((size_t)preO[l] * d.Kpad * 2);
    unsigned short* outF = (unsigned short*)alloc((size_t)Bn * preO[l] * d.Ho * d.Wo * 2);
    unsigned short* outL = (unsigned short*)alloc((size_t)Bn * preO[l] * d.Ho * d.Wo * 2);
    run_conv(stream, curF, pFw[l], pFb[l], wpF, nullptr, outF, preI[l], H, W, preO[l], 3, preS[l], 1, 1);
    run_conv(stream, curL, pLw[l], pLb[l], wpL, nullptr, outL, preI[l], H, W, preO[l], 3, preS[l], 1, 1);
    curF = outF; curL = outL; H = d.Ho; W = d.Wo;
  }

  // --- pyramid levels ---
  const int SPLITc[4] = {160, 320, 640, 640};
  const int INJ[4]    = {320, 640, 1280, 1280};
  const int RESL[4]   = {64, 32, 16, 8};
  size_t out_off = 0;
  int Cin = 64;
  for (int i = 0; i < 4; ++i) {
    const int res = RESL[i];
    const int C   = SPLITc[i];
    const int hw  = res * res;
    ConvDims de = conv_dims(H, W, Cin, 3, 2, 1);  // de.Ho == res

    // ext convs (SiLU), keep fp32 for metric/splat and bf16 for next level
    unsigned short* wpF    = (unsigned short*)alloc((size_t)C * de.Kpad * 2);
    unsigned short* wpL    = (unsigned short*)alloc((size_t)C * de.Kpad * 2);
    float*          featF  = (float*)alloc((size_t)Bn * C * hw * 4);
    float*          featL  = (float*)alloc((size_t)Bn * C * hw * 4);
    unsigned short* featFb = (unsigned short*)alloc((size_t)Bn * C * hw * 2);
    unsigned short* featLb = (unsigned short*)alloc((size_t)Bn * C * hw * 2);
    run_conv(stream, curF, eFw[i], eFb[i], wpF, featF, featFb, Cin, H, W, C, 3, 2, 1, 1);
    run_conv(stream, curL, eLw[i], eLb[i], wpL, featL, featLb, Cin, H, W, C, 3, 2, 1, 1);

    // flow pyramid
    float* flF = (float*)alloc((size_t)Bn * 2 * hw * 4);
    float* flB = (float*)alloc((size_t)Bn * 2 * hw * 4);
    {
      int n = Bn * 2 * hw;
      flow_resize_kernel<<<CDIV(n, 256), 256, 0, stream>>>(flow, flF, 0, res);
      flow_resize_kernel<<<CDIV(n, 256), 256, 0, stream>>>(flow, flB, 2, res);
    }

    // occlusion masks
    float* occF = (float*)alloc((size_t)Bn * hw * 4);
    float* occB = (float*)alloc((size_t)Bn * hw * 4);
    {
      int n = Bn * hw;
      occ_mask_kernel<<<CDIV(n, 256), 256, 0, stream>>>(flF, flB, occF, res);
      occ_mask_kernel<<<CDIV(n, 256), 256, 0, stream>>>(flB, flF, occB, res);
    }

    // metric -> Z
    float* Zf = (float*)alloc((size_t)Bn * hw * 4);
    float* Zb = (float*)alloc((size_t)Bn * hw * 4);
    {
      int n = Bn * hw;
      metric_z_kernel<<<CDIV(n, 256), 256, 0, stream>>>(featF, mw[i], mb[i], occF, Zf, C, res);
      metric_z_kernel<<<CDIV(n, 256), 256, 0, stream>>>(featL, mw[i], mb[i], occB, Zb, C, res);
    }

    // splat accumulators (zeroed every call -> deterministic)
    float* accF = (float*)alloc((size_t)Bn * (C + 1) * hw * 4);
    float* accB = (float*)alloc((size_t)Bn * (C + 1) * hw * 4);
    {
      int n = Bn * (C + 1) * hw;
      zero_f32_kernel<<<CDIV(n, 256), 256, 0, stream>>>(accF, n);
      zero_f32_kernel<<<CDIV(n, 256), 256, 0, stream>>>(accB, n);
      splat_kernel<<<CDIV(n, 256), 256, 0, stream>>>(featF, Zf, flF, accF, C, res);
      splat_kernel<<<CDIV(n, 256), 256, 0, stream>>>(featL, Zb, flB, accB, C, res);
    }

    // fuse
    unsigned short* fusedb = (unsigned short*)alloc((size_t)Bn * C * hw * 2);
    {
      int n = Bn * C * hw;
      fuse_kernel<<<CDIV(n, 256), 256, 0, stream>>>(accF, accB, occF, occB, fusedb, C, res);
    }

    // zero conv (no activation) -> d_out slice
    ConvDims dz = conv_dims(res, res, C, 3, 1, 1);
    unsigned short* wpZ = (unsigned short*)alloc((size_t)INJ[i] * dz.Kpad * 2);
    run_conv(stream, fusedb, zw[i], zb[i], wpZ, (float*)d_out + out_off, nullptr,
             C, res, res, INJ[i], 3, 1, 1, 0);
    out_off += (size_t)Bn * INJ[i] * hw;

    // next level consumes this level's bf16 features
    Cin = C; H = res; W = res; curF = featFb; curL = featLb;
  }
}
